// FactorizedRandomAttention_45337674777146
// MI455X (gfx1250) — matrix operand order repaired and checked by reading, not compiled, not run on hardware
//
#include <hip/hip_runtime.h>

typedef __attribute__((ext_vector_type(16))) _Float16 v16h;
typedef __attribute__((ext_vector_type(8)))  float    v8f;

namespace {
constexpr int Sdim    = 4096;   // sequence length
constexpr int Fr      = 64;     // factor rank (K of score GEMM)
constexpr int Dh      = 64;     // head dim
constexpr int Hn      = 8;      // heads
constexpr int NC      = 32;     // column chunk (== K of PV WMMA)
constexpr int WAVES   = 8;      // waves per workgroup
constexpr int ROWS_WG = WAVES * 16;     // 128 rows / workgroup
constexpr int NCHUNK  = Sdim / NC;      // 128 column chunks
}

__global__ __launch_bounds__(WAVES * 32)
void fra_attn_kernel(const float* __restrict__ v,
                     const float* __restrict__ A1,
                     const float* __restrict__ A2,
                     const int*   __restrict__ mask,
                     float* __restrict__ out,
                     float* __restrict__ attn)
{
    // LDS staging buffers (f16), shared by all 8 waves of the WG
    __shared__ _Float16 sB[NC][Fr + 4];        // A2 chunk, [n][k]
    __shared__ _Float16 sV[Dh][NC + 4];        // v  chunk, [d][k]
    __shared__ _Float16 sP[WAVES][16][NC];     // per-wave P tile bounce [m][n]

    const int tid  = threadIdx.x;
    const int wave = tid >> 5;
    const int lane = tid & 31;
    const int l16  = lane & 15;
    const int hi   = (lane >> 4) & 1;
    const int hi8  = hi << 3;

    const int h  = blockIdx.x / (Sdim / ROWS_WG);
    const int rb = blockIdx.x % (Sdim / ROWS_WG);
    const int mw = rb * ROWS_WG + wave * 16;   // this wave's first row

    // ---- A1 operand: 16x64 rows of this wave, f32->f16, WMMA A layout ----
    // A 16x32 layout: lane(0..15)=M ; halves: g<4 -> K=2g+r, g>=4 -> K=2g+8+r ;
    // lanes 16..31 add +8 to K. Two K-chunks (0..31, 32..63).
    const float* A1w = A1 + ((size_t)h * Sdim + (mw + l16)) * Fr;
    v16h a_lo, a_hi;
#pragma unroll
    for (int e = 0; e < 16; ++e) {
        const int g = e >> 1, r = e & 1;
        const int k = 2 * g + r + ((g >= 4) ? 8 : 0) + (hi ? 8 : 0);
        a_lo[e] = (_Float16)A1w[k];
        a_hi[e] = (_Float16)A1w[k + 32];
    }

    // =================== PASS 1: row sums of exp(score) ===================
    float rsum[8];
#pragma unroll
    for (int i = 0; i < 8; ++i) rsum[i] = 0.f;

    for (int cc = 0; cc < NCHUNK; ++cc) {
        const int nc0 = cc * NC;
        {   // stage A2 chunk -> sB[n][k]   (64 x 32 f32, 8 elems/thread)
            const int k  = tid >> 2;
            const int nb = (tid & 3) * 8;
            const float* src = A2 + ((size_t)h * Fr + k) * Sdim + nc0 + nb;
#pragma unroll
            for (int j = 0; j < 8; ++j) sB[nb + j][k] = (_Float16)src[j];
            if (cc + 1 < NCHUNK)   // pull next chunk toward L2 (global_prefetch_b8)
                __builtin_prefetch(src + NC, 0, 3);
        }
        __syncthreads();

#pragma unroll
        for (int t = 0; t < 2; ++t) {
            // B operand 32x16: g<4 -> K=2g+r, g>=4 -> K=2g+8+r; lanes 16..31 add +8 to K, N=lane&15
            v16h b_lo, b_hi;
#pragma unroll
            for (int e = 0; e < 16; ++e) {
                const int g = e >> 1, r = e & 1;
                const int k = 2 * g + r + ((g >= 4) ? 8 : 0) + (hi ? 8 : 0);
                b_lo[e] = sB[t * 16 + l16][k];
                b_hi[e] = sB[t * 16 + l16][k + 32];
            }
            v8f c = {};
            c = __builtin_amdgcn_wmma_f32_16x16x32_f16(false, a_lo, false, b_lo,
                                                       (short)0, c, false, false);
            c = __builtin_amdgcn_wmma_f32_16x16x32_f16(false, a_hi, false, b_hi,
                                                       (short)0, c, false, false);
            const int col = nc0 + t * 16 + l16;
#pragma unroll
            for (int i = 0; i < 8; ++i) {
                const int row = mw + i + hi8;
                const int mk  = mask[(size_t)row * Sdim + col];
                rsum[i] += (mk != 0) ? 0.f : __expf(c[i]);
            }
        }
        __syncthreads();
    }

    // reduce row sums across the 16 lanes holding the same row
    float rinv[8];
#pragma unroll
    for (int i = 0; i < 8; ++i) {
        float s = rsum[i];
        s += __shfl_xor(s, 1);  s += __shfl_xor(s, 2);
        s += __shfl_xor(s, 4);  s += __shfl_xor(s, 8);
        rinv[i] = 1.0f / s;
    }

    // ========= PASS 2: write normalized attn + accumulate O = P·V =========
    v8f oc[4] = {v8f{}, v8f{}, v8f{}, v8f{}};

    for (int cc = 0; cc < NCHUNK; ++cc) {
        const int nc0 = cc * NC;
        {   // stage A2 chunk -> sB[n][k]
            const int k  = tid >> 2;
            const int nb = (tid & 3) * 8;
            const float* src = A2 + ((size_t)h * Fr + k) * Sdim + nc0 + nb;
#pragma unroll
            for (int j = 0; j < 8; ++j) sB[nb + j][k] = (_Float16)src[j];
        }
        {   // stage v chunk -> sV[d][k]    (32 x 64 f32, 8 elems/thread)
            const int k  = tid >> 3;
            const int db = (tid & 7) * 8;
            const float* src = v + ((size_t)h * Sdim + nc0 + k) * Dh + db;
#pragma unroll
            for (int j = 0; j < 8; ++j) sV[db + j][k] = (_Float16)src[j];
            if (cc + 1 < NCHUNK)
                __builtin_prefetch(src + (size_t)NC * Dh, 0, 3);
        }
        __syncthreads();

#pragma unroll
        for (int t = 0; t < 2; ++t) {
            v16h b_lo, b_hi;
#pragma unroll
            for (int e = 0; e < 16; ++e) {
                const int g = e >> 1, r = e & 1;
                const int k = 2 * g + r + ((g >= 4) ? 8 : 0) + (hi ? 8 : 0);
                b_lo[e] = sB[t * 16 + l16][k];
                b_hi[e] = sB[t * 16 + l16][k + 32];
            }
            v8f c = {};
            c = __builtin_amdgcn_wmma_f32_16x16x32_f16(false, a_lo, false, b_lo,
                                                       (short)0, c, false, false);
            c = __builtin_amdgcn_wmma_f32_16x16x32_f16(false, a_hi, false, b_hi,
                                                       (short)0, c, false, false);
            const int col = nc0 + t * 16 + l16;
#pragma unroll
            for (int i = 0; i < 8; ++i) {
                const int row = mw + i + hi8;
                const int mk  = mask[(size_t)row * Sdim + col];
                const float p = (mk != 0) ? 0.f : __expf(c[i]) * rinv[i];
                attn[((size_t)h * Sdim + row) * Sdim + col] = p;
                sP[wave][i + hi8][t * 16 + l16] = (_Float16)p;   // C-layout -> LDS
            }
        }

        // gather P tile (16x32) back in WMMA A layout (wave-private, DScnt-ordered)
        v16h pa;
#pragma unroll
        for (int e = 0; e < 16; ++e) {
            const int g = e >> 1, r = e & 1;
            const int k = 2 * g + r + ((g >= 4) ? 8 : 0) + (hi ? 8 : 0);
            pa[e] = sP[wave][l16][k];
        }
        // O(16x64) += P(16x32) * V(32x64), four N-subtiles
#pragma unroll
        for (int dt = 0; dt < 4; ++dt) {
            v16h bv;
#pragma unroll
            for (int e = 0; e < 16; ++e) {
                const int g = e >> 1, r = e & 1;
                const int k = 2 * g + r + ((g >= 4) ? 8 : 0) + (hi ? 8 : 0);
                bv[e] = sV[dt * 16 + l16][k];
            }
            oc[dt] = __builtin_amdgcn_wmma_f32_16x16x32_f16(false, pa, false, bv,
                                                            (short)0, oc[dt], false, false);
        }
        __syncthreads();
    }

    // ---- write O (already normalized) ----
#pragma unroll
    for (int dt = 0; dt < 4; ++dt)
#pragma unroll
        for (int i = 0; i < 8; ++i)
            out[((size_t)h * Sdim + (mw + i + hi8)) * Dh + dt * 16 + l16] = oc[dt][i];
}

extern "C" void kernel_launch(void* const* d_in, const int* in_sizes, int n_in,
                              void* d_out, int out_size, void* d_ws, size_t ws_size,
                              hipStream_t stream) {
    (void)in_sizes; (void)n_in; (void)d_ws; (void)ws_size; (void)out_size;
    const float* v    = (const float*)d_in[0];
    const float* A1   = (const float*)d_in[1];
    const float* A2   = (const float*)d_in[2];
    const int*   mask = (const int*)d_in[3];
    // d_in[4] = len_q (== Sdim for this problem size)

    float* out  = (float*)d_out;                                  // (1,8,4096,64)
    float* attn = (float*)d_out + (size_t)Hn * Sdim * Dh;         // (1,8,4096,4096)

    dim3 grid(Hn * (Sdim / ROWS_WG));   // 8 heads * 32 row blocks = 256 WGs
    dim3 block(WAVES * 32);             // 256 threads = 8 wave32 waves
    hipLaunchKernelGGL(fra_attn_kernel, grid, block, 0, stream,
                       v, A1, A2, mask, out, attn);
}
